// ErdosGATConvGraphGymLayer_88072599371911
// MI455X (gfx1250) — matrix operand order repaired and checked by reading, not compiled, not run on hardware
//
#include <hip/hip_runtime.h>
#include <hip/hip_bf16.h>
#include <math.h>

typedef __attribute__((ext_vector_type(16))) _Float16 v16h;
typedef __attribute__((ext_vector_type(8)))  _Float16 v8h;
typedef __attribute__((ext_vector_type(8)))  float    v8f;

#define N_NODES   50000
#define F_DIM     64
#define H_HEADS   4
#define D_DIM     64
#define E_EDGES   800000
#define G_GRAPHS  64
#define C_DIM     256          // H*D
#define EN_TOT    (E_EDGES + N_NODES)   // edges + self loops
#define NEG_SLOPE 0.2f

// ---------------------------------------------------------------- utilities

__global__ void k_fill_f32(float* __restrict__ p, float v, int n) {
    int i = blockIdx.x * blockDim.x + threadIdx.x;
    int stride = gridDim.x * blockDim.x;
    for (; i < n; i += stride) p[i] = v;
}

__global__ void k_cvt_f16(const float* __restrict__ in, _Float16* __restrict__ out, int n) {
    int i = blockIdx.x * blockDim.x + threadIdx.x;
    if (i < n) out[i] = (_Float16)in[i];
}

__device__ __forceinline__ void atomicMaxF(float* addr, float v) {
    // sign-aware float max via int/uint atomics (works with -inf init)
    if (v >= 0.f) atomicMax((int*)addr, __float_as_int(v));
    else          atomicMin((unsigned int*)addr, (unsigned int)__float_as_int(v));
}

// A-fragment loader: 16x32 f16, lane holds row m, K halves per ISA layout
__device__ __forceinline__ v16h load_a_frag(const _Float16* __restrict__ rowptr,
                                            int k0, int asub) {
    v8h lo = *(const v8h*)(rowptr + k0 + asub);        // K = k0+asub .. +7
    v8h hi = *(const v8h*)(rowptr + k0 + 16 + asub);   // K = k0+16+asub .. +7
    v16h r;
#pragma unroll
    for (int i = 0; i < 8; ++i) { r[i] = lo[i]; r[i + 8] = hi[i]; }
    return r;
}

// ---------------------------------------------------------------- mask scatter
// msum[row] += mask[col]  (thresholded later when consumed)
__global__ void k_mask_scatter(const float* __restrict__ mask,
                               const long long* __restrict__ ei,
                               float* __restrict__ msum) {
    int tid = blockIdx.x * blockDim.x + threadIdx.x;   // E*16 threads (float4 each)
    if (tid >= E_EDGES * 16) return;
    int e = tid >> 4, q = tid & 15;
    long long r = ei[e];
    long long c = ei[(long long)E_EDGES + e];
    const float4 mv = *(const float4*)(mask + c * F_DIM + q * 4);
    float* dst = msum + r * F_DIM + q * 4;
    atomicAdd(dst + 0, mv.x);
    atomicAdd(dst + 1, mv.y);
    atomicAdd(dst + 2, mv.z);
    atomicAdd(dst + 3, mv.w);
}

// ---------------------------------------------------------------- GEMM1 (WMMA)
// xw[N,256] = x[N,64] @ W_gat^T ; both K-contiguous row-major (NT form)
__global__ void k_gemm1_wmma(const _Float16* __restrict__ xh,
                             const _Float16* __restrict__ wh,
                             float* __restrict__ xw) {
    int wave = (blockIdx.x * blockDim.x + threadIdx.x) >> 5;
    int lane = threadIdx.x & 31;
    int rowTile = wave >> 3;          // 8 waves per 16-row tile
    int wsub    = wave & 7;           // each wave: 2 column tiles
    if (rowTile >= N_NODES / 16) return;
    int m0 = rowTile * 16;

    const _Float16* arow = xh + (size_t)(m0 + (lane & 15)) * 64;
    int asub = (lane < 16) ? 0 : 8;
    v16h a0 = load_a_frag(arow, 0, asub);
    v16h a1 = load_a_frag(arow, 32, asub);
    int bsub = (lane < 16) ? 0 : 8;

#pragma unroll
    for (int t = 0; t < 2; ++t) {
        int n0 = (wsub * 2 + t) * 16;
        const _Float16* bcol = wh + (size_t)(n0 + (lane & 15)) * 64;
        v16h b0 = load_a_frag(bcol, 0, bsub);
        v16h b1 = load_a_frag(bcol, 32, bsub);
        v8f acc = {};
        acc = __builtin_amdgcn_wmma_f32_16x16x32_f16(false, a0, false, b0,
                                                     (short)0, acc, false, false);
        acc = __builtin_amdgcn_wmma_f32_16x16x32_f16(false, a1, false, b1,
                                                     (short)0, acc, false, false);
        int col = n0 + (lane & 15);
        int mb  = m0 + ((lane < 16) ? 0 : 8);
        float* out = xw + (size_t)mb * C_DIM + col;
#pragma unroll
        for (int r = 0; r < 8; ++r) out[(size_t)r * C_DIM] = acc[r];
    }
}

// ---------------------------------------------------------------- attention dots
__global__ void k_att_dot(const float* __restrict__ xw,
                          const float* __restrict__ att_src,
                          const float* __restrict__ att_dst,
                          float* __restrict__ a_s, float* __restrict__ a_d) {
    int idx = blockIdx.x * blockDim.x + threadIdx.x;   // N*H
    if (idx >= N_NODES * H_HEADS) return;
    int i = idx >> 2, h = idx & 3;
    const float4* xp = (const float4*)(xw + (size_t)i * C_DIM + h * D_DIM);
    const float4* sp = (const float4*)(att_src + h * D_DIM);
    const float4* dp = (const float4*)(att_dst + h * D_DIM);
    float s = 0.f, d = 0.f;
#pragma unroll
    for (int k = 0; k < 16; ++k) {
        float4 xv = xp[k], sv = sp[k], dv = dp[k];
        s += xv.x * sv.x + xv.y * sv.y + xv.z * sv.z + xv.w * sv.w;
        d += xv.x * dv.x + xv.y * dv.y + xv.z * dv.z + xv.w * dv.w;
    }
    a_s[idx] = s; a_d[idx] = d;
}

// ---------------------------------------------------------------- edge passes
__device__ __forceinline__ void edge_sd(int e, const long long* __restrict__ ei,
                                        long long& s, long long& d) {
    if (e < E_EDGES) { s = ei[e]; d = ei[(long long)E_EDGES + e]; }
    else             { s = d = e - E_EDGES; }       // self loop
}

__global__ void k_edge_max(const long long* __restrict__ ei,
                           const float* __restrict__ a_s, const float* __restrict__ a_d,
                           float* __restrict__ emax) {
    int tid = blockIdx.x * blockDim.x + threadIdx.x;   // EN*H
    if (tid >= EN_TOT * H_HEADS) return;
    int e = tid >> 2, h = tid & 3;
    long long s, d; edge_sd(e, ei, s, d);
    float v = a_s[s * 4 + h] + a_d[d * 4 + h];
    v = (v > 0.f) ? v : NEG_SLOPE * v;
    atomicMaxF(&emax[d * 4 + h], v);
}

__global__ void k_edge_exp(const long long* __restrict__ ei,
                           const float* __restrict__ a_s, const float* __restrict__ a_d,
                           const float* __restrict__ emax,
                           float* __restrict__ exbuf, float* __restrict__ denom) {
    int tid = blockIdx.x * blockDim.x + threadIdx.x;   // EN*H
    if (tid >= EN_TOT * H_HEADS) return;
    int e = tid >> 2, h = tid & 3;
    long long s, d; edge_sd(e, ei, s, d);
    float v = a_s[s * 4 + h] + a_d[d * 4 + h];
    v = (v > 0.f) ? v : NEG_SLOPE * v;
    float ex = __expf(v - emax[d * 4 + h]);
    exbuf[tid] = ex;
    atomicAdd(&denom[d * 4 + h], ex);
}

__global__ void k_edge_scatter(const long long* __restrict__ ei,
                               const float* __restrict__ exbuf,
                               const float* __restrict__ denom,
                               const float* __restrict__ xw,
                               float* __restrict__ outg) {
    long long tid = (long long)blockIdx.x * blockDim.x + threadIdx.x; // EN*64
    if (tid >= (long long)EN_TOT * 64) return;
    int e = (int)(tid >> 6);
    int q = (int)(tid & 63);      // float4 chunk over 256 columns
    int h = q >> 4;
    long long s, d; edge_sd(e, ei, s, d);
    float alpha = exbuf[(size_t)e * 4 + h] / (denom[d * 4 + h] + 1e-16f);
    float4 xv = *(const float4*)(xw + s * C_DIM + q * 4);
    float* dst = outg + d * C_DIM + q * 4;
    atomicAdd(dst + 0, alpha * xv.x);
    atomicAdd(dst + 1, alpha * xv.y);
    atomicAdd(dst + 2, alpha * xv.z);
    atomicAdd(dst + 3, alpha * xv.w);
}

// ---------------------------------------------------------------- graph degree
__global__ void k_deg(const long long* __restrict__ batch, float* __restrict__ deg) {
    int i = blockIdx.x * blockDim.x + threadIdx.x;
    if (i < N_NODES) atomicAdd(&deg[batch[i]], 1.0f);
}

// ---------------------------------------------------------------- fused hidden
// hid[i,c] = f16( (outg[i,c] + bias[c]) * (msum[i, c&63] > 0 ? 1 : 0) )
__global__ void k_hidden(const float* __restrict__ outg, const float* __restrict__ bias,
                         const float* __restrict__ msum, _Float16* __restrict__ hid) {
    int tid = blockIdx.x * blockDim.x + threadIdx.x;   // N*64 (float4 each)
    if (tid >= N_NODES * 64) return;
    int i = tid >> 6, q = tid & 63;
    int c = q * 4;
    float4 g = *(const float4*)(outg + (size_t)i * C_DIM + c);
    float4 b = *(const float4*)(bias + c);
    float4 m = *(const float4*)(msum + (size_t)i * F_DIM + (c & 63));
    _Float16* dst = hid + (size_t)i * C_DIM + c;
    dst[0] = (_Float16)((g.x + b.x) * (m.x > 0.f ? 1.f : 0.f));
    dst[1] = (_Float16)((g.y + b.y) * (m.y > 0.f ? 1.f : 0.f));
    dst[2] = (_Float16)((g.z + b.z) * (m.z > 0.f ? 1.f : 0.f));
    dst[3] = (_Float16)((g.w + b.w) * (m.w > 0.f ? 1.f : 0.f));
}

// ---------------------------------------------------------------- GEMM2 (WMMA)
// out[N,64] = hid[N,256] @ W_lin^T, fused +b_lin and GraphSizeNorm epilogue
__global__ void k_gemm2_wmma(const _Float16* __restrict__ hid,
                             const _Float16* __restrict__ wl,
                             const float* __restrict__ b_lin,
                             const long long* __restrict__ batch,
                             const float* __restrict__ deg,
                             float* __restrict__ out) {
    int wave = (blockIdx.x * blockDim.x + threadIdx.x) >> 5;
    int lane = threadIdx.x & 31;
    if (wave >= (N_NODES / 16) * 4) return;
    int rowTile = wave >> 2, ct = wave & 3;
    int m0 = rowTile * 16, n0 = ct * 16;

    const _Float16* arow = hid + (size_t)(m0 + (lane & 15)) * C_DIM;
    int asub = (lane < 16) ? 0 : 8;
    const _Float16* bcol = wl + (size_t)(n0 + (lane & 15)) * C_DIM;
    int bsub = (lane < 16) ? 0 : 8;
    __builtin_prefetch(bcol, 0, 3);

    v8f acc = {};
#pragma unroll
    for (int k0 = 0; k0 < C_DIM; k0 += 32) {
        v16h a = load_a_frag(arow, k0, asub);
        v16h b = load_a_frag(bcol, k0, bsub);
        acc = __builtin_amdgcn_wmma_f32_16x16x32_f16(false, a, false, b,
                                                     (short)0, acc, false, false);
    }

    int col = n0 + (lane & 15);
    float bb = b_lin[col];
    int mb = m0 + ((lane < 16) ? 0 : 8);
#pragma unroll
    for (int r = 0; r < 8; ++r) {
        int row = mb + r;
        float dg = deg[batch[row]];
        float inv = rsqrtf(dg > 0.f ? dg : 1.f);
        out[(size_t)row * D_DIM + col] = (acc[r] + bb) * inv;
    }
}

// ---------------------------------------------------------------- launch

extern "C" void kernel_launch(void* const* d_in, const int* in_sizes, int n_in,
                              void* d_out, int out_size, void* d_ws, size_t ws_size,
                              hipStream_t stream) {
    const float*     x        = (const float*)d_in[0];
    const float*     mask     = (const float*)d_in[1];
    const long long* ei       = (const long long*)d_in[2];
    const long long* batch    = (const long long*)d_in[3];
    const float*     W_gat    = (const float*)d_in[4];
    const float*     att_src  = (const float*)d_in[5];
    const float*     att_dst  = (const float*)d_in[6];
    const float*     bias_gat = (const float*)d_in[7];
    const float*     W_lin    = (const float*)d_in[8];
    const float*     b_lin    = (const float*)d_in[9];
    float*           out      = (float*)d_out;

    char* ws = (char*)d_ws;
    size_t off = 0;
    auto carve = [&](size_t bytes) -> void* {
        void* p = ws + off;
        off += (bytes + 255) & ~(size_t)255;
        return p;
    };
    float*     msum   = (float*)carve((size_t)N_NODES * F_DIM * 4);
    float*     xw     = (float*)carve((size_t)N_NODES * C_DIM * 4);
    float*     a_s    = (float*)carve((size_t)N_NODES * H_HEADS * 4);
    float*     a_d    = (float*)carve((size_t)N_NODES * H_HEADS * 4);
    float*     emax   = (float*)carve((size_t)N_NODES * H_HEADS * 4);
    float*     denom  = (float*)carve((size_t)N_NODES * H_HEADS * 4);
    float*     exbuf  = (float*)carve((size_t)EN_TOT * H_HEADS * 4);
    float*     outg   = (float*)carve((size_t)N_NODES * C_DIM * 4);
    _Float16*  hid    = (_Float16*)carve((size_t)N_NODES * C_DIM * 2);
    _Float16*  xh     = (_Float16*)carve((size_t)N_NODES * F_DIM * 2);
    _Float16*  wgat_h = (_Float16*)carve((size_t)C_DIM * F_DIM * 2);
    _Float16*  wlin_h = (_Float16*)carve((size_t)D_DIM * C_DIM * 2);
    float*     deg    = (float*)carve((size_t)G_GRAPHS * 4);

    const int B = 256;
    const float NEG_INF = -__builtin_huge_valf();

    // init accumulators (every call; ws is not re-zeroed by harness)
    k_fill_f32<<<2048, B, 0, stream>>>(msum, 0.f, N_NODES * F_DIM);
    k_fill_f32<<<4096, B, 0, stream>>>(outg, 0.f, N_NODES * C_DIM);
    k_fill_f32<<<512,  B, 0, stream>>>(emax, NEG_INF, N_NODES * H_HEADS);
    k_fill_f32<<<512,  B, 0, stream>>>(denom, 0.f, N_NODES * H_HEADS);
    k_fill_f32<<<1,    64, 0, stream>>>(deg, 0.f, G_GRAPHS);

    // f32 -> f16 conversions for WMMA inputs
    k_cvt_f16<<<(N_NODES * F_DIM + B - 1) / B, B, 0, stream>>>(x, xh, N_NODES * F_DIM);
    k_cvt_f16<<<(C_DIM * F_DIM + B - 1) / B, B, 0, stream>>>(W_gat, wgat_h, C_DIM * F_DIM);
    k_cvt_f16<<<(D_DIM * C_DIM + B - 1) / B, B, 0, stream>>>(W_lin, wlin_h, D_DIM * C_DIM);

    // mask segment-sum scatter
    k_mask_scatter<<<(E_EDGES * 16 + B - 1) / B, B, 0, stream>>>(mask, ei, msum);

    // xw = x @ W_gat^T  (WMMA)
    k_gemm1_wmma<<<N_NODES / 16, B, 0, stream>>>(xh, wgat_h, xw);

    // attention logits
    k_att_dot<<<(N_NODES * H_HEADS + B - 1) / B, B, 0, stream>>>(xw, att_src, att_dst, a_s, a_d);

    // segment softmax + weighted scatter
    int enh = EN_TOT * H_HEADS;
    k_edge_max<<<(enh + B - 1) / B, B, 0, stream>>>(ei, a_s, a_d, emax);
    k_edge_exp<<<(enh + B - 1) / B, B, 0, stream>>>(ei, a_s, a_d, emax, exbuf, denom);
    long long sc = (long long)EN_TOT * 64;
    k_edge_scatter<<<(int)((sc + B - 1) / B), B, 0, stream>>>(ei, exbuf, denom, xw, outg);

    // graph sizes
    k_deg<<<(N_NODES + B - 1) / B, B, 0, stream>>>(batch, deg);

    // hid = f16((outg + bias) * (msum > 0))
    k_hidden<<<(N_NODES * 64 + B - 1) / B, B, 0, stream>>>(outg, bias_gat, msum, hid);

    // out = (hid @ W_lin^T + b_lin) * inv_sqrt(deg[batch])  (WMMA + fused epilogue)
    int tiles = (N_NODES / 16) * 4;
    k_gemm2_wmma<<<(tiles + 7) / 8, B, 0, stream>>>(hid, wlin_h, b_lin, batch, deg, out);
}
